// Model_27462020891077
// MI455X (gfx1250) — compile-verified, matrix operand order checked by reading, not run on hardware
//
#include <hip/hip_runtime.h>
#include <hip/hip_bf16.h>
#include <math.h>

#define DIM        2048
#define VOCAB      128000
#define NTOK       2048
#define SOFTCAP_F  30.0f
#define NCHUNK     125        // 125 chunks * 1024 cols = 128000
#define COLS_PER_CHUNK 1024   // 8 passes * 8 waves * 16 cols
#define PASSES     8

// LDS row stride for the A panel: 4096 data bytes + 4*16B TDM pads = 4160 B.
// (pad every 1024 B of data, pad_amount = 4 dwords -> keeps 16B alignment,
//  spreads lanes over banks instead of stride-4096 all-same-bank.)
#define A_ROW_STRIDE 4160

typedef __bf16 v16bf __attribute__((ext_vector_type(16)));
typedef __bf16 v8bf  __attribute__((ext_vector_type(8)));
typedef float  v8f   __attribute__((ext_vector_type(8)));
typedef unsigned int u32x4 __attribute__((ext_vector_type(4)));
typedef int          i32x8 __attribute__((ext_vector_type(8)));
typedef int          i32x4 __attribute__((ext_vector_type(4)));

// ---------------------------------------------------------------------------
// Kernel 1: fused bf16 WMMA GEMM + softcap + online softmax partials.
// Block = 256 threads (8 waves). blockIdx.y = token tile (16 tokens),
// blockIdx.x = vocab chunk (1024 columns). The 16x2048 A panel (x tile) is
// DMA'd into LDS once per workgroup via the Tensor Data Mover; B (W rows)
// streams straight from HBM to VGPRs. Each wave computes a 16x16 logits tile
// per pass via v_wmma_f32_16x16x32_bf16, K-loop over 2048.
// ---------------------------------------------------------------------------
__global__ __launch_bounds__(256)
void ce_partial_kernel(const __hip_bfloat16* __restrict__ x,
                       const __hip_bfloat16* __restrict__ w,
                       float2* __restrict__ partials)
{
    const int lane = threadIdx.x & 31;
    const int wave = threadIdx.x >> 5;
    const int half = lane >> 4;   // 0: lanes 0-15, 1: lanes 16-31
    const int r    = lane & 15;

    const int tok_base = blockIdx.y << 4;
    const int chunk    = blockIdx.x;

    __shared__ __align__(16) unsigned char lds_a[16 * A_ROW_STRIDE];
    __shared__ float lm[8][16];
    __shared__ float ls[8][16];

    // ---- TDM: DMA the 16x2048 bf16 A panel into LDS (once per workgroup) ---
    if (threadIdx.x < 32) {                     // wave 0 issues the DMA
        unsigned long long ga =
            (unsigned long long)(uintptr_t)((const __bf16*)x +
                                            (size_t)tok_base * DIM);
        unsigned int lds_off = (unsigned int)(uintptr_t)&lds_a[0];

        u32x4 g0;
        g0.x = 1u;                                      // count=1, user mode
        g0.y = lds_off;                                 // lds_addr
        g0.z = (unsigned int)ga;                        // global_addr[31:0]
        g0.w = (unsigned int)((ga >> 32) & 0x01FFFFFFu) // global_addr[56:32]
             | (2u << 30);                              // type=2 ("image")

        i32x8 g1;
        g1.s0 = (int)(0x00010000u      // data_size=1 -> 2 bytes
                    | (1u << 20)       // pad_enable
                    | (7u << 22)       // pad_interval: every 1024 B
                    | (3u << 25));     // pad_amount: 4 dwords (16 B)
        g1.s1 = (int)(2048u << 16);    // tensor_dim0 = 2048 (lo16)
        g1.s2 = (int)(16u   << 16);    // tensor_dim0 hi=0 | tensor_dim1 = 16
        g1.s3 = (int)(2048u << 16);    // tensor_dim1 hi=0 | tile_dim0 = 2048
        g1.s4 = 16;                    // tile_dim1 = 16, tile_dim2 = 0
        g1.s5 = 2048;                  // tensor_dim0_stride = 2048 elements
        g1.s6 = 0;
        g1.s7 = 0;

        i32x4 z4 = {0, 0, 0, 0};       // groups 2/3 unused (2D tensor)
        i32x8 z8 = {0, 0, 0, 0, 0, 0, 0, 0};
        __builtin_amdgcn_tensor_load_to_lds(g0, g1, z4, z4, z8, 0);
        __builtin_amdgcn_s_wait_tensorcnt(0);
    }
    __syncthreads();

    // Per-lane K sub-offsets per ISA 16-bit A/B VGPR layout:
    // lanes 0-15: K = {k..k+7, k+16..k+23}; lanes 16-31: K = {k+8..k+15, k+24..k+31}
    const int k_off0 = half * 8;
    const int k_off1 = 16 + half * 8;

    // A-panel LDS base for this lane's token row (row r of the tile).
    const unsigned rb = (unsigned)r * A_ROW_STRIDE;

    v8f m_run, s_run;
    #pragma unroll
    for (int i = 0; i < 8; ++i) { m_run[i] = -INFINITY; s_run[i] = 0.f; }

    #pragma unroll 1
    for (int pass = 0; pass < PASSES; ++pass) {
        const int col = chunk * COLS_PER_CHUNK + pass * 128 + wave * 16 + r;
        const __bf16* wrow = (const __bf16*)w + (size_t)col * DIM;

        v8f c;
        #pragma unroll
        for (int i = 0; i < 8; ++i) c[i] = 0.f;

        #pragma unroll 4
        for (int k = 0; k < DIM; k += 32) {
            // Prefetch ahead in the W stream (dominant HBM traffic).
            __builtin_prefetch(wrow + k + 512, 0, 1);

            // A from LDS (padded layout: +16 B per 1024 B of row data).
            unsigned e0 = (unsigned)(k + k_off0) * 2u;
            unsigned e1 = (unsigned)(k + k_off1) * 2u;
            v8bf a0 = *(const v8bf*)(lds_a + rb + e0 + ((e0 >> 10) << 4));
            v8bf a1 = *(const v8bf*)(lds_a + rb + e1 + ((e1 >> 10) << 4));
            // B straight from global (the 524 MB W stream).
            v8bf b0 = *(const v8bf*)(wrow + k + k_off0);
            v8bf b1 = *(const v8bf*)(wrow + k + k_off1);

            v16bf va = __builtin_shufflevector(a0, a1,
                         0,1,2,3,4,5,6,7,8,9,10,11,12,13,14,15);
            v16bf vb = __builtin_shufflevector(b0, b1,
                         0,1,2,3,4,5,6,7,8,9,10,11,12,13,14,15);
            c = __builtin_amdgcn_wmma_f32_16x16x32_bf16(
                    /*neg_a=*/false, va, /*neg_b=*/false, vb,
                    /*c_mod=*/(short)0, c,
                    /*reuse_a=*/false, /*reuse_b=*/false);
        }

        // Softcap + elementwise online-softmax update (per lane: 8 tokens,
        // one vocab column of this tile).
        #pragma unroll
        for (int i = 0; i < 8; ++i) {
            float l  = SOFTCAP_F * tanhf(c[i] * (1.0f / SOFTCAP_F));
            float mn = fmaxf(m_run[i], l);
            s_run[i] = s_run[i] * __expf(m_run[i] - mn) + __expf(l - mn);
            m_run[i] = mn;
        }
    }

    // Reduce (m, s) across the 16 columns held by the 16 lanes of each half.
    #pragma unroll
    for (int off = 8; off >= 1; off >>= 1) {
        #pragma unroll
        for (int i = 0; i < 8; ++i) {
            float mo = __shfl_xor(m_run[i], off, 32);
            float so = __shfl_xor(s_run[i], off, 32);
            float mn = fmaxf(m_run[i], mo);
            s_run[i] = s_run[i] * __expf(m_run[i] - mn)
                     + so        * __expf(mo       - mn);
            m_run[i] = mn;
        }
    }

    // C-layout rows: lanes 0-15 VGPR i -> token i; lanes 16-31 -> token i+8.
    if (r == 0) {
        #pragma unroll
        for (int i = 0; i < 8; ++i) {
            int tok = i + half * 8;
            lm[wave][tok] = m_run[i];
            ls[wave][tok] = s_run[i];
        }
    }
    __syncthreads();

    // Merge across the 8 waves; one (m,s) partial per (token, chunk).
    if (threadIdx.x < 16) {
        int tok = threadIdx.x;
        float M = -INFINITY, S = 0.f;
        #pragma unroll
        for (int wv = 0; wv < 8; ++wv) {
            float mo = lm[wv][tok], so = ls[wv][tok];
            float mn = fmaxf(M, mo);
            S = S * __expf(M - mn) + so * __expf(mo - mn);
            M = mn;
        }
        partials[(size_t)(tok_base + tok) * NCHUNK + chunk] = make_float2(M, S);
    }
}

// ---------------------------------------------------------------------------
// Kernel 2: per-token finalize. Block per token: target-logit dot product
// (256 threads * 8 elems = 2048 dims) + log-sum-exp merge of 125 partials.
// ---------------------------------------------------------------------------
__global__ __launch_bounds__(256)
void ce_finalize_kernel(const __hip_bfloat16* __restrict__ x,
                        const int* __restrict__ targets,
                        const __hip_bfloat16* __restrict__ w,
                        const float2* __restrict__ partials,
                        float* __restrict__ out)
{
    __shared__ float red[8];
    __shared__ float pm[NCHUNK];
    __shared__ float ps[NCHUNK];

    const int t = blockIdx.x;
    int tgt = targets[t];
    if (tgt < 0) tgt = 0;
    if (tgt >= VOCAB) tgt = VOCAB - 1;

    const __bf16* xr = (const __bf16*)x + (size_t)t   * DIM;
    const __bf16* wr = (const __bf16*)w + (size_t)tgt * DIM;

    const int d0 = threadIdx.x * 8;
    v8bf xv = *(const v8bf*)(xr + d0);
    v8bf wv = *(const v8bf*)(wr + d0);
    float acc = 0.f;
    #pragma unroll
    for (int i = 0; i < 8; ++i) acc += (float)xv[i] * (float)wv[i];

    #pragma unroll
    for (int off = 16; off >= 1; off >>= 1)
        acc += __shfl_xor(acc, off, 32);
    if ((threadIdx.x & 31) == 0) red[threadIdx.x >> 5] = acc;

    if (threadIdx.x < NCHUNK) {
        float2 p = partials[(size_t)t * NCHUNK + threadIdx.x];
        pm[threadIdx.x] = p.x;
        ps[threadIdx.x] = p.y;
    }
    __syncthreads();

    if (threadIdx.x == 0) {
        float dot = 0.f;
        #pragma unroll
        for (int i = 0; i < 8; ++i) dot += red[i];
        float ltgt = SOFTCAP_F * tanhf(dot * (1.0f / SOFTCAP_F));

        float M = -INFINITY;
        for (int i = 0; i < NCHUNK; ++i) M = fmaxf(M, pm[i]);
        float S = 0.f;
        for (int i = 0; i < NCHUNK; ++i) S += ps[i] * __expf(pm[i] - M);

        out[t] = M + __logf(S) - ltgt;   // -log p[target]
    }
}

// ---------------------------------------------------------------------------
extern "C" void kernel_launch(void* const* d_in, const int* in_sizes, int n_in,
                              void* d_out, int out_size, void* d_ws, size_t ws_size,
                              hipStream_t stream)
{
    (void)in_sizes; (void)n_in; (void)out_size; (void)ws_size;

    const __hip_bfloat16* x       = (const __hip_bfloat16*)d_in[0];
    const int*            targets = (const int*)d_in[1];
    const __hip_bfloat16* weight  = (const __hip_bfloat16*)d_in[2];
    float*                out     = (float*)d_out;
    float2*               partials = (float2*)d_ws;   // NTOK * NCHUNK * 8 B ~= 2 MB

    dim3 grid1(NCHUNK, NTOK / 16);
    ce_partial_kernel<<<grid1, 256, 0, stream>>>(x, weight, partials);
    ce_finalize_kernel<<<NTOK, 256, 0, stream>>>(x, targets, weight, partials, out);
}
